// RotAttentionPool2d_29738353557865
// MI455X (gfx1250) — matrix operand order repaired and checked by reading, not compiled, not run on hardware
//
#include <hip/hip_runtime.h>

// ---------------------------------------------------------------------------
// RotAttentionPool2d, CLS-token-only output.  B=64, C=1024, N=196+1 tokens,
// HEADS=16, HEAD_DIM=64, REF=7.  Only out[:,0] is needed, so:
//   - q only for CLS (no RoPE on CLS)
//   - K,V for all 197 tokens: (B*197,1024)x(1024,2048) bf16 WMMA GEMM (dominant)
//   - tiny softmax-attention row per (b,head), then (64,1024)x(1024,1024) proj.
// KV GEMM: M flattened across batch (12608 = 197 blocks of 64, no guards),
// block tile 64x128, wave tile 32x32 (4 WMMAs / k-step). A tile double-buffered
// in LDS via global_load_async_to_lds_b128 (ASYNCcnt), B fragments direct from
// L2-resident N-major weights.
// ---------------------------------------------------------------------------

#define BB    64
#define CC    1024
#define NTOK  196
#define TOK   197
#define E3    3072
#define NKV   2048
#define MTOT  (BB * TOK)          // 12608 flattened rows

typedef __attribute__((ext_vector_type(16))) __bf16 v16bf;
typedef __attribute__((ext_vector_type(8)))  __bf16 v8bf;
typedef __attribute__((ext_vector_type(8)))  float  v8f;

__device__ __forceinline__ unsigned short f2bf(float f) {
  unsigned u = __builtin_bit_cast(unsigned, f);
  unsigned r = u + 0x7FFFu + ((u >> 16) & 1u);   // round-to-nearest-even
  return (unsigned short)(r >> 16);
}
__device__ __forceinline__ float bf2f(unsigned short h) {
  return __builtin_bit_cast(float, (unsigned)h << 16);
}

// A fragment (16x32 bf16, MxK), row pointer pre-resolved per lane.
__device__ __forceinline__ v16bf frag_a(const unsigned short* rowp, int hi) {
  union { v16bf v; v8bf h[2]; } u;
  u.h[0] = *(const v8bf*)(rowp + hi * 16);
  u.h[1] = *(const v8bf*)(rowp + hi * 16 + 8);
  return u.v;
}
// B fragment (32x16 bf16, KxN) from N-major storage: lane holds N row,
// elements j -> K = k0 + hi*16 + j (contiguous 16 bf16).
__device__ __forceinline__ v16bf frag_b(const unsigned short* nrowp, int hi) {
  union { v16bf v; v8bf h[2]; } u;
  const unsigned short* p = nrowp + hi * 16;
  u.h[0] = *(const v8bf*)(p);
  u.h[1] = *(const v8bf*)(p + 8);
  return u.v;
}

#define WMMA_BF16(A, Bm, Cacc) \
  __builtin_amdgcn_wmma_f32_16x16x32_bf16(false, (A), false, (Bm), (short)0, (Cacc), false, false)

__device__ __forceinline__ void async_b128_to_lds(unsigned lds_addr, const void* gaddr) {
  asm volatile("global_load_async_to_lds_b128 %0, %1, off"
               :: "v"(lds_addr), "v"(gaddr) : "memory");
}
__device__ __forceinline__ void wait_async0() {
  asm volatile("s_wait_asynccnt 0x0" ::: "memory");
}

// ---- 1) weight transpose + bf16 convert: Wt[n][k] = W[k][col_off+n] ----------
__global__ void wT_kernel(const float* __restrict__ W, int ld, int col_off,
                          unsigned short* __restrict__ Wt, int Kdim) {
  __shared__ float tile[32][33];
  int n0 = blockIdx.x * 32, k0 = blockIdx.y * 32;
  int tx = threadIdx.x, ty = threadIdx.y;
  tile[ty][tx] = W[(size_t)(k0 + ty) * ld + col_off + n0 + tx];
  __syncthreads();
  Wt[(size_t)(n0 + ty) * Kdim + k0 + tx] = f2bf(tile[tx][ty]);
}

// ---- 2) x transpose to token-major bf16: T[b][1+t][c] = x[b][c][t] -----------
__global__ void xpose_kernel(const float* __restrict__ x,
                             unsigned short* __restrict__ T) {
  __shared__ float tile[32][33];
  int b = blockIdx.z;
  int t0 = blockIdx.y * 32;
  int c0 = blockIdx.x * 32;
  int tx = threadIdx.x, ty = threadIdx.y;
  if (t0 + tx < NTOK)
    tile[ty][tx] = x[((size_t)b * CC + c0 + ty) * NTOK + t0 + tx];
  __syncthreads();
  int tw = t0 + ty, cw = c0 + tx;
  if (tw < NTOK)
    T[((size_t)b * TOK + 1 + tw) * CC + cw] = f2bf(tile[tx][ty]);
}

// ---- 3) per-(b,c) mean over 196 tokens; one wave per (b,c) -------------------
__global__ void mean_kernel(const float* __restrict__ x,
                            float* __restrict__ mean,
                            unsigned short* __restrict__ T) {
  int wave = (int)((blockIdx.x * blockDim.x + threadIdx.x) >> 5);
  int lane = threadIdx.x & 31;
  if (wave >= BB * CC) return;
  const float* p = x + (size_t)wave * NTOK;
  float s = 0.f;
  for (int i = lane; i < NTOK; i += 32) s += p[i];
  #pragma unroll
  for (int off = 16; off; off >>= 1) s += __shfl_down(s, off, 32);
  if (lane == 0) {
    float m = s * (1.0f / (float)NTOK);
    mean[wave] = m;
    int b = wave >> 10, c = wave & 1023;
    T[(size_t)b * TOK * CC + c] = f2bf(m);   // CLS token row
  }
}

// ---- 4) q_cls = mean @ Wq + bq  (f32; 0.13 GFLOP) ----------------------------
__global__ void qcls_kernel(const float* __restrict__ mean,
                            const float* __restrict__ qkv_w,
                            const float* __restrict__ qkv_b,
                            float* __restrict__ q) {
  int e = blockIdx.x * blockDim.x + threadIdx.x;   // 0..1023
  int b = blockIdx.y;
  const float* m = mean + (size_t)b * CC;
  float acc = qkv_b[e];
  for (int c = 0; c < CC; ++c) acc += m[c] * qkv_w[(size_t)c * E3 + e];
  q[(size_t)b * CC + e] = acc;
}

// ---- helper: biased bf16 store routed to K or V ------------------------------
__device__ __forceinline__ void store_kv(unsigned short* __restrict__ Kout,
                                         unsigned short* __restrict__ Vout,
                                         const float* __restrict__ qkv_b,
                                         size_t row, int n, float acc) {
  float v = acc + qkv_b[CC + n];
  if (n < CC) Kout[row + n]      = f2bf(v);
  else        Vout[row + n - CC] = f2bf(v);
}

// ---- 5) main WMMA GEMM: KV = T @ Wkv^T ---------------------------------------
// grid (NKV/128 = 16, MTOT/64 = 197); block 256 = 8 waves.
// Block tile 64(M) x 128(N); wave tile 32x32 (2 A frags x 2 B frags, 4 WMMAs).
// A tile (64 x 32k) double-buffered in LDS via async loads; rows padded to 40
// bf16 (80B) for bank-conflict-free 16-row fragment reads.
#define AP 40
__global__ void __launch_bounds__(256)
kv_gemm_kernel(const unsigned short* __restrict__ T,   // (12608,1024) bf16
               const unsigned short* __restrict__ Wt,  // (2048,1024)  bf16 N-major
               const float* __restrict__ qkv_b,
               unsigned short* __restrict__ Kout,      // (12608,1024) bf16
               unsigned short* __restrict__ Vout) {    // (12608,1024) bf16
  __shared__ unsigned short sA[2][64 * AP];
  int nblk = blockIdx.x * 128;
  int mblk = blockIdx.y * 64;
  int tid  = threadIdx.x;
  int w = tid >> 5, lane = tid & 31;
  int l = lane & 15, hi = lane >> 4;
  int wm = w >> 2, wn = w & 3;
  int n0  = nblk + wn * 32;
  int am0 = wm * 32;                       // wave's local M base

  // async copy mapping: one b128 (8 bf16) per thread per 32-wide k slab
  int arow   = tid >> 2;                   // 0..63
  int achunk = (tid & 3) * 8;              // 0,8,16,24
  const unsigned short* gA = T + (size_t)(mblk + arow) * CC + achunk;
  unsigned ldsA0 = (unsigned)(size_t)&sA[0][arow * AP + achunk];
  unsigned ldsA1 = (unsigned)(size_t)&sA[1][arow * AP + achunk];

  const unsigned short* Brow0 = Wt + (size_t)(n0 + l) * CC;
  const unsigned short* Brow1 = Wt + (size_t)(n0 + 16 + l) * CC;

  v8f acc00 = {}, acc01 = {}, acc10 = {}, acc11 = {};

  async_b128_to_lds(ldsA0, gA);            // prologue: slab 0 -> buf 0

  for (int ks = 0; ks < CC / 32; ++ks) {
    wait_async0();
    __syncthreads();                       // buf[cur] fully written, visible
    if (ks + 1 < CC / 32) {
      unsigned dst = (ks & 1) ? ldsA0 : ldsA1;
      async_b128_to_lds(dst, gA + (ks + 1) * 32);
    }
    int k0 = ks * 32;
    __builtin_prefetch(Brow0 + k0 + 128, 0, 0);  // global_prefetch_b8
    __builtin_prefetch(Brow1 + k0 + 128, 0, 0);
    const unsigned short* base = &sA[ks & 1][0];
    v16bf a0 = frag_a(base + (am0 + l) * AP, hi);        // ds_load_b128
    v16bf a1 = frag_a(base + (am0 + 16 + l) * AP, hi);
    v16bf b0 = frag_b(Brow0 + k0, hi);                   // global_load_b128
    v16bf b1 = frag_b(Brow1 + k0, hi);
    acc00 = WMMA_BF16(a0, b0, acc00);
    acc01 = WMMA_BF16(a0, b1, acc01);
    acc10 = WMMA_BF16(a1, b0, acc10);
    acc11 = WMMA_BF16(a1, b1, acc11);
    __syncthreads();                       // all waves done reading buf[cur]
  }

  int nA = n0 + l, nB = n0 + 16 + l;
  #pragma unroll
  for (int r = 0; r < 8; ++r) {
    size_t row0 = (size_t)(mblk + am0 + r + hi * 8) * CC;
    size_t row1 = (size_t)(mblk + am0 + 16 + r + hi * 8) * CC;
    store_kv(Kout, Vout, qkv_b, row0, nA, acc00[r]);
    store_kv(Kout, Vout, qkv_b, row0, nB, acc01[r]);
    store_kv(Kout, Vout, qkv_b, row1, nA, acc10[r]);
    store_kv(Kout, Vout, qkv_b, row1, nB, acc11[r]);
  }
}

// ---- 6) RoPE on K spatial tokens (bf16 in/out) -------------------------------
__global__ void rope_kernel(unsigned short* __restrict__ Kmat) {
  size_t idx = (size_t)blockIdx.x * blockDim.x + threadIdx.x;
  if (idx >= (size_t)BB * NTOK * (CC / 2)) return;
  int pr = (int)(idx & 511);
  size_t bt = idx >> 9;
  int p = (int)(bt % NTOK);
  int b = (int)(bt / NTOK);
  int e0 = pr * 2;
  int di = (e0 & 63) >> 1;          // 0..31
  int axis = di >> 4;               // 0 -> y, 1 -> x
  int band = di & 15;
  float theta = __powf(10000.0f, -(float)band * (1.0f / 16.0f));
  float coord = axis ? (float)(p % 14) : (float)(p / 14);
  float ang = coord * 0.5f * theta;           // coord/14 * REF(=7)
  float s, c;
  __sincosf(ang, &s, &c);
  size_t base = ((size_t)b * TOK + 1 + p) * CC + e0;
  float a = bf2f(Kmat[base]), bb = bf2f(Kmat[base + 1]);
  Kmat[base]     = f2bf(a * c - bb * s);
  Kmat[base + 1] = f2bf(bb * c + a * s);
}

// ---- 7) CLS attention per (b, head) ------------------------------------------
__global__ void __launch_bounds__(256)
attn_kernel(const float* __restrict__ q, const unsigned short* __restrict__ K,
            const unsigned short* __restrict__ V, unsigned short* __restrict__ Obf) {
  __shared__ float sc[200];
  __shared__ float red[8];
  int b = blockIdx.x, h = blockIdx.y;
  int tid = threadIdx.x;
  const float* qh = q + (size_t)b * CC + h * 64;
  float s = -1e30f;
  if (tid < TOK) {
    const unsigned short* kh = K + ((size_t)b * TOK + tid) * CC + h * 64;
    float acc = 0.f;
    #pragma unroll 8
    for (int d = 0; d < 64; ++d) acc += qh[d] * bf2f(kh[d]);
    s = acc * 0.125f;                 // HEAD_DIM^-0.5
    sc[tid] = s;
  }
  float m = s;
  #pragma unroll
  for (int off = 16; off; off >>= 1) m = fmaxf(m, __shfl_down(m, off, 32));
  if ((tid & 31) == 0) red[tid >> 5] = m;
  __syncthreads();
  float mall = fmaxf(fmaxf(fmaxf(red[0], red[1]), fmaxf(red[2], red[3])),
                     fmaxf(fmaxf(red[4], red[5]), fmaxf(red[6], red[7])));
  float e = 0.f;
  if (tid < TOK) { e = __expf(sc[tid] - mall); sc[tid] = e; }
  float t = e;
  #pragma unroll
  for (int off = 16; off; off >>= 1) t += __shfl_down(t, off, 32);
  __syncthreads();
  if ((tid & 31) == 0) red[tid >> 5] = t;
  __syncthreads();
  float sum = (red[0] + red[1]) + (red[2] + red[3]) +
              (red[4] + red[5]) + (red[6] + red[7]);
  float inv = 1.0f / sum;
  if (tid < 64) {
    const unsigned short* vb = V + (size_t)b * TOK * CC + h * 64 + tid;
    float acc = 0.f;
    for (int tt = 0; tt < TOK; ++tt) acc += sc[tt] * bf2f(vb[(size_t)tt * CC]);
    Obf[(size_t)b * CC + h * 64 + tid] = f2bf(acc * inv);
  }
}

// ---- 8) output projection: out = O @ proj_w + proj_b (WMMA) ------------------
__global__ void __launch_bounds__(256)
proj_gemm_kernel(const unsigned short* __restrict__ A,
                 const unsigned short* __restrict__ Pt,
                 const float* __restrict__ pb, float* __restrict__ out) {
  int nblk = blockIdx.x * 64;
  int w = threadIdx.x >> 5, lane = threadIdx.x & 31;
  int l = lane & 15, hi = lane >> 4;
  int m0 = (w & 3) * 16;
  int n0 = nblk + (w >> 2) * 32;
  const unsigned short* Arow  = A  + (size_t)(m0 + l) * CC;
  const unsigned short* Brow0 = Pt + (size_t)(n0 + l) * CC;
  const unsigned short* Brow1 = Pt + (size_t)(n0 + 16 + l) * CC;
  v8f acc0 = {}, acc1 = {};
  for (int k0 = 0; k0 < CC; k0 += 32) {
    v16bf a  = frag_a(Arow  + k0, hi);
    v16bf b0 = frag_b(Brow0 + k0, hi);
    v16bf b1 = frag_b(Brow1 + k0, hi);
    acc0 = WMMA_BF16(a, b0, acc0);
    acc1 = WMMA_BF16(a, b1, acc1);
  }
  #pragma unroll
  for (int r = 0; r < 8; ++r) {
    int m = m0 + r + hi * 8;
    int n = n0 + l;
    out[(size_t)m * CC + n]      = acc0[r] + pb[n];
    out[(size_t)m * CC + n + 16] = acc1[r] + pb[n + 16];
  }
}

// ---------------------------------------------------------------------------
extern "C" void kernel_launch(void* const* d_in, const int* in_sizes, int n_in,
                              void* d_out, int out_size, void* d_ws, size_t ws_size,
                              hipStream_t stream) {
  const float* x      = (const float*)d_in[0];
  const float* qkv_w  = (const float*)d_in[1];
  const float* qkv_b  = (const float*)d_in[2];
  const float* proj_w = (const float*)d_in[3];
  const float* proj_b = (const float*)d_in[4];
  float* out = (float*)d_out;
  char* ws = (char*)d_ws;

  size_t offT    = 0;                               // bf16 (12608,1024)
  size_t szT     = (size_t)MTOT * CC * 2;
  size_t offWt   = offT + szT;                      // bf16 (2048,1024)
  size_t szWt    = (size_t)NKV * CC * 2;
  size_t offPt   = offWt + szWt;                    // bf16 (1024,1024)
  size_t szPt    = (size_t)CC * CC * 2;
  size_t offMean = offPt + szPt;                    // f32 (B,1024)
  size_t offQ    = offMean + (size_t)BB * CC * 4;   // f32 (B,1024)
  size_t offK    = offQ + (size_t)BB * CC * 4;      // bf16 (12608,1024)
  size_t szKV    = (size_t)MTOT * CC * 2;
  size_t offV    = offK + szKV;                     // bf16 (12608,1024)
  size_t offObf  = offV + szKV;                     // bf16 (B,1024)

  unsigned short* T    = (unsigned short*)(ws + offT);
  unsigned short* Wt   = (unsigned short*)(ws + offWt);
  unsigned short* Pt   = (unsigned short*)(ws + offPt);
  float*          Mean = (float*)(ws + offMean);
  float*          Q    = (float*)(ws + offQ);
  unsigned short* Km   = (unsigned short*)(ws + offK);
  unsigned short* Vm   = (unsigned short*)(ws + offV);
  unsigned short* Obf  = (unsigned short*)(ws + offObf);

  wT_kernel<<<dim3(NKV / 32, CC / 32), dim3(32, 32), 0, stream>>>(qkv_w, E3, CC, Wt, CC);
  wT_kernel<<<dim3(CC / 32, CC / 32),  dim3(32, 32), 0, stream>>>(proj_w, CC, 0, Pt, CC);

  xpose_kernel<<<dim3(CC / 32, (NTOK + 31) / 32, BB), dim3(32, 32), 0, stream>>>(x, T);
  mean_kernel<<<(BB * CC) / 8, 256, 0, stream>>>(x, Mean, T);
  qcls_kernel<<<dim3(CC / 256, BB), 256, 0, stream>>>(Mean, qkv_w, qkv_b, Q);

  kv_gemm_kernel<<<dim3(NKV / 128, MTOT / 64), 256, 0, stream>>>(T, Wt, qkv_b, Km, Vm);

  {
    size_t total = (size_t)BB * NTOK * (CC / 2);
    rope_kernel<<<(unsigned)((total + 255) / 256), 256, 0, stream>>>(Km);
  }

  attn_kernel<<<dim3(BB, 16), 256, 0, stream>>>(Q, Km, Vm, Obf);
  proj_gemm_kernel<<<dim3(CC / 64), 256, 0, stream>>>(Obf, Pt, proj_b, out);

  (void)in_sizes; (void)n_in; (void)out_size; (void)ws_size;
}
